// MambaDehaze_14577119002795
// MI455X (gfx1250) — matrix operand order repaired and checked by reading, not compiled, not run on hardware
//
#include <hip/hip_runtime.h>
#include <cstdint>
#include <cstddef>

// ---------------- CDNA5 WMMA types ----------------
typedef __attribute__((ext_vector_type(16))) _Float16 v16h;
typedef __attribute__((ext_vector_type(8)))  float    v8f;

#define DEVFN __device__ __forceinline__

DEVFN float siluf(float v) { return v / (1.f + __expf(-v)); }
DEVFN float geluf(float v) { return 0.5f * v * (1.f + erff(v * 0.7071067811865476f)); }
DEVFN float softplusf(float v) { return (v > 20.f) ? v : log1pf(__expf(v)); }

namespace cfg {
constexpr int IMG = 512, PS = 8, ED = 64, DST = 16, WSZ = 8, NLAY = 4, NB = 2, DI = 128;
constexpr int GRD = IMG / PS;          // 64 patch grid
constexpr int NP  = GRD * GRD;         // 4096 patches/img
constexpr int NT  = NB * NP;           // 8192 tokens total
constexpr int NWIN = NT / (WSZ * WSZ); // 128 windows
constexpr int KF1 = 9 * 192;           // 1728 : fus1 implicit-GEMM K
}

// ---------------- generic f32->f16 cast with zero padding ----------------
__global__ void k_cast_pad2(const float* __restrict__ src, _Float16* __restrict__ dst,
                            int srows, int scols, int drows, int dcols) {
  int i = blockIdx.x * 256 + threadIdx.x;
  int total = drows * dcols;
  if (i >= total) return;
  int r = i / dcols, c = i % dcols;
  float v = (r < srows && c < scols) ? src[r * scols + c] : 0.f;
  dst[i] = (_Float16)v;
}

// ---------------- patch-embed im2col: hazy (B,3,512,512) -> A[8192][192] ----------------
__global__ void k_im2col(const float* __restrict__ hazy, _Float16* __restrict__ A) {
  int i = blockIdx.x * 256 + threadIdx.x;
  if (i >= cfg::NT * 192) return;
  int t = i / 192, k = i % 192;
  int b = t >> 12, p = t & 4095, gy = p >> 6, gx = p & 63;
  int c = k >> 6, r = (k >> 3) & 7, q = k & 7;
  A[i] = (_Float16)hazy[(((size_t)(b * 3 + c) * 512) + gy * 8 + r) * 512 + gx * 8 + q];
}

__global__ void k_add_pos(float* __restrict__ x, const float* __restrict__ pos) {
  int i = blockIdx.x * 256 + threadIdx.x;
  if (i >= cfg::NT * cfg::ED) return;
  x[i] += pos[i & (cfg::NP * cfg::ED - 1)];
}

// ---------------- window partition (with cyclic shift) into f16 GEMM A-matrix ----------------
__global__ void k_win_part(const float* __restrict__ x, _Float16* __restrict__ aw, int shift) {
  int i = blockIdx.x * 256 + threadIdx.x;
  if (i >= cfg::NT * cfg::ED) return;
  int wt = i >> 6, c = i & 63;
  int b = wt >> 12, rem = wt & 4095;
  int win = rem >> 6, l = rem & 63;
  int wyb = win >> 3, wxb = win & 7, wy = l >> 3, wx = l & 7;
  int sy = ((wyb << 3) + wy + shift) & 63;
  int sx = ((wxb << 3) + wx + shift) & 63;
  aw[i] = (_Float16)x[(((size_t)(b << 12) + (sy << 6) + sx) << 6) + c];
}

// ---------------- WMMA GEMM:  C[M][N] = A[M][K] * W[N][K]^T (+bias, act) ----------------
// A row-major f16, W row-major (N-major, K contiguous) f16 == exactly the jax
// Linear weight layout. One wave per 16x16 tile; block of 4 waves covers 64 rows.
// Fragments use contiguous k = 16*(lane/16) + j in A and B (16-bit A 16x32, B 32x16, f32 C/D).
__global__ __launch_bounds__(128) void k_gemm(
    const _Float16* __restrict__ A, int lda,
    const _Float16* __restrict__ W, int ldb,
    const float* __restrict__ bias,
    float* __restrict__ Cf, _Float16* __restrict__ Ch, int ldc,
    int M, int N, int K, int act) {
  int lane = threadIdx.x & 31;
  int wave = threadIdx.x >> 5;
  int n0 = blockIdx.x * 16;
  int m0 = blockIdx.y * 64 + wave * 16;
  if (m0 >= M) return;

  int mrow = lane & 15;
  int hi   = lane >> 4;       // 0 or 1
  int kba  = hi << 3;         // unused (0 / 8); A loads use kbb
  int kbb  = hi << 4;         // B half-chunk base (0 / 16)
  const _Float16* arow = A + (size_t)(m0 + mrow) * lda;
  const _Float16* brow = W + (size_t)(n0 + mrow) * ldb;   // n = lane&15 for B frag

  v8f acc = {};
  for (int k0 = 0; k0 < K; k0 += 32) {
    v16h af, bf;
    const _Float16* ap = arow + k0;
#pragma unroll
    for (int j = 0; j < 16; ++j) af[j] = ap[kbb + j];
    const _Float16* bp = brow + k0 + kbb;
#pragma unroll
    for (int j = 0; j < 16; ++j) bf[j] = bp[j];
    acc = __builtin_amdgcn_wmma_f32_16x16x32_f16(false, af, false, bf,
                                                 (short)0, acc, false, false);
  }

  int n  = n0 + (lane & 15);
  int mb = m0 + (hi << 3);
  float bv = bias ? bias[n] : 0.f;
#pragma unroll
  for (int r = 0; r < 8; ++r) {
    float v = acc[r] + bv;
    if (act == 1) v = siluf(v);
    else if (act == 2) v = geluf(v);
    size_t o = (size_t)(mb + r) * ldc + n;
    if (Cf) Cf[o] = v;
    if (Ch) Ch[o] = (_Float16)v;
  }
}

// ---------------- depthwise causal conv (k=4) + SiLU, directional ----------------
// xz f16 [NT][256] (u part = cols 0..127). u out f16 [NT][128].
__global__ void k_dwconv(const _Float16* __restrict__ xz,
                         const float* __restrict__ cw, const float* __restrict__ cb,
                         _Float16* __restrict__ u, int dir) {
  int idx = blockIdx.x * 256 + threadIdx.x;
  if (idx >= cfg::NWIN * cfg::DI) return;
  int win = idx >> 7, c = idx & 127;
  float w0 = cw[c * 4 + 0], w1 = cw[c * 4 + 1], w2 = cw[c * 4 + 2], w3 = cw[c * 4 + 3];
  float bb = cb[c];
  float h0 = 0.f, h1 = 0.f, h2 = 0.f;
  for (int i = 0; i < 64; ++i) {
    int l = (dir > 0) ? i : 63 - i;
    size_t t = (size_t)win * 64 + l;
    float xv = (float)xz[t * 256 + c];
    float s = w0 * h0 + w1 * h1 + w2 * h2 + w3 * xv + bb;
    u[t * 128 + c] = (_Float16)siluf(s);
    h0 = h1; h1 = h2; h2 = xv;
  }
}

// ---------------- selective scan (per window, per channel), directional ----------------
// dbl f32 [NT][48]: cols 0..3 dt_in, 4..19 B, 20..35 C.  Gated output -> y f16 [NT][128].
__global__ void k_scan(const float* __restrict__ dbl, const _Float16* __restrict__ u,
                       const _Float16* __restrict__ xz,
                       const float* __restrict__ dtw, const float* __restrict__ dtb,
                       const float* __restrict__ Alog, const float* __restrict__ Dp,
                       _Float16* __restrict__ yo, int dir) {
  int idx = blockIdx.x * 256 + threadIdx.x;
  if (idx >= cfg::NWIN * cfg::DI) return;
  int win = idx >> 7, c = idx & 127;
  float q0 = dtw[c * 4 + 0], q1 = dtw[c * 4 + 1], q2 = dtw[c * 4 + 2], q3 = dtw[c * 4 + 3];
  float db = dtb[c], Dv = Dp[c];
  float Av[16], h[16];
#pragma unroll
  for (int s = 0; s < 16; ++s) { Av[s] = -__expf(Alog[c * 16 + s]); h[s] = 0.f; }
  for (int i = 0; i < 64; ++i) {
    int l = (dir > 0) ? i : 63 - i;
    size_t t = (size_t)win * 64 + l;
    const float* d = dbl + t * 48;
    float dt = softplusf(q0 * d[0] + q1 * d[1] + q2 * d[2] + q3 * d[3] + db);
    float uv = (float)u[t * 128 + c];
    float y = 0.f;
#pragma unroll
    for (int s = 0; s < 16; ++s) {
      float dA = __expf(dt * Av[s]);
      h[s] = dA * h[s] + dt * d[4 + s] * uv;
      y += h[s] * d[20 + s];
    }
    float zv = (float)xz[t * 256 + 128 + c];
    yo[t * 128 + c] = (_Float16)((y + uv * Dv) * siluf(zv));
  }
}

// ---------------- un-window (+shift back) + residual + LayerNorm ----------------
__global__ void k_unwin_ln(const float* __restrict__ xin, const float* __restrict__ yfus,
                           const float* __restrict__ g, const float* __restrict__ bln,
                           float* __restrict__ xout, int shift) {
  int t = blockIdx.x * 256 + threadIdx.x;
  if (t >= cfg::NT) return;
  int b = t >> 12, p = t & 4095, sy = p >> 6, sx = p & 63;
  int r = (sy - shift) & 63, rc = (sx - shift) & 63;
  int wt = ((b << 6) + ((r >> 3) << 3) + (rc >> 3)) * 64 + ((r & 7) << 3) + (rc & 7);
  const float* xi = xin + (size_t)t * 64;
  const float* yf = yfus + (size_t)wt * 64;
  float sum = 0.f, sq = 0.f;
  for (int c = 0; c < 64; ++c) { float v = xi[c] + yf[c]; sum += v; sq += v * v; }
  float m = sum * (1.f / 64.f);
  float var = sq * (1.f / 64.f) - m * m;
  float inv = rsqrtf(var + 1e-5f);
  for (int c = 0; c < 64; ++c) {
    float v = xi[c] + yf[c];
    xout[(size_t)t * 64 + c] = (v - m) * inv * g[c] + bln[c];
  }
}

// ---------------- small direct conv3x3 (NHWC, pad 1) + GELU ----------------
__global__ void k_conv3x3(const float* __restrict__ in, const float* __restrict__ w,
                          const float* __restrict__ bias, float* __restrict__ out,
                          int Bn, int H, int W, int Ci, int Co, int stride) {
  int Ho = H / stride, Wo = W / stride;
  size_t total = (size_t)Bn * Ho * Wo * Co;
  size_t idx = (size_t)blockIdx.x * 256 + threadIdx.x;
  if (idx >= total) return;
  int o = idx % Co;
  int x = (idx / Co) % Wo;
  int y = (idx / ((size_t)Co * Wo)) % Ho;
  int b = idx / ((size_t)Co * Wo * Ho);
  float s = bias[o];
  for (int ky = 0; ky < 3; ++ky) {
    int iy = y * stride + ky - 1;
    if ((unsigned)iy >= (unsigned)H) continue;
    for (int kx = 0; kx < 3; ++kx) {
      int ix = x * stride + kx - 1;
      if ((unsigned)ix >= (unsigned)W) continue;
      const float* ip = in + ((size_t)(b * H + iy) * W + ix) * Ci;
      const float* wp = w + (((size_t)o * Ci) * 3 + ky) * 3 + kx;
      for (int ci = 0; ci < Ci; ++ci) s += ip[ci] * wp[(size_t)ci * 9];
    }
  }
  out[idx] = geluf(s);
}

// ---------------- transposed conv 2x2 stride 2 (einsum bchw,codk->bohdwk) + GELU ----------------
__global__ void k_convT2(const float* __restrict__ in, const float* __restrict__ w,
                         const float* __restrict__ bias, float* __restrict__ out,
                         int Bn, int H, int W, int Ci, int Co) {
  size_t total = (size_t)Bn * 4 * H * W * Co;
  size_t idx = (size_t)blockIdx.x * 256 + threadIdx.x;
  if (idx >= total) return;
  int o = idx % Co;
  int x = (idx / Co) % (2 * W);
  int y = (idx / ((size_t)Co * 2 * W)) % (2 * H);
  int b = idx / ((size_t)Co * 4 * H * W);
  int hh = y >> 1, d = y & 1, ww = x >> 1, kk = x & 1;
  float s = bias[o];
  const float* ip = in + ((size_t)(b * H + hh) * W + ww) * Ci;
  for (int c = 0; c < Ci; ++c) s += ip[c] * w[(((size_t)c * Co + o) * 2 + d) * 2 + kk];
  out[idx] = geluf(s);
}

// ---------------- bilinear upsample (half-pixel) into padded f16 concat (B,514,514,192) ----------------
__global__ void k_bilinear(const float* __restrict__ in, int Bn, int H, int W, int C,
                           _Float16* __restrict__ catp, int co) {
  size_t total = (size_t)Bn * 512 * 512 * C;
  size_t idx = (size_t)blockIdx.x * 256 + threadIdx.x;
  if (idx >= total) return;
  int c = idx % C;
  int x = (idx / C) % 512;
  int y = (idx / ((size_t)C * 512)) % 512;
  int b = idx / ((size_t)C * 512 * 512);
  float sy = (y + 0.5f) * (float)H / 512.f - 0.5f;
  float sx = (x + 0.5f) * (float)W / 512.f - 0.5f;
  float fy0 = floorf(sy), fx0 = floorf(sx);
  int y0 = (int)fy0, x0 = (int)fx0;
  float fy = sy - fy0, fx = sx - fx0;
  if (y0 < 0) { y0 = 0; fy = 0.f; }
  if (x0 < 0) { x0 = 0; fx = 0.f; }
  int y1 = min(y0 + 1, H - 1), x1 = min(x0 + 1, W - 1);
  if (y0 > H - 1) y0 = H - 1;
  if (x0 > W - 1) x0 = W - 1;
  const float* base = in + (size_t)b * H * W * C;
  float v00 = base[((size_t)y0 * W + x0) * C + c];
  float v01 = base[((size_t)y0 * W + x1) * C + c];
  float v10 = base[((size_t)y1 * W + x0) * C + c];
  float v11 = base[((size_t)y1 * W + x1) * C + c];
  float v = (1.f - fy) * ((1.f - fx) * v00 + fx * v01) + fy * ((1.f - fx) * v10 + fx * v11);
  catp[(((size_t)b * 514 + (y + 1)) * 514 + (x + 1)) * 192 + co + c] = (_Float16)v;
}

// ---------------- repack fus1_w (64,192,3,3) -> (64)[tap*192+c] f16 ----------------
__global__ void k_pack_fus1(const float* __restrict__ src, _Float16* __restrict__ dst) {
  int i = blockIdx.x * 256 + threadIdx.x;
  if (i >= 64 * cfg::KF1) return;
  int o = i / cfg::KF1, kk = i % cfg::KF1;
  int tap = kk / 192, c = kk % 192, ky = tap / 3, kx = tap % 3;
  dst[i] = (_Float16)src[(((size_t)o * 192 + c) * 3 + ky) * 3 + kx];
}

// ---------------- fusion conv as implicit GEMM with WMMA + LDS halo staging ----------------
// One block = 64 consecutive pixels of one image row. The 3x66x192 f16 input halo
// (76 KB, fits CDNA5's 320 KB/WGP LDS) is staged once and reused by all 9 taps.
// Each wave computes 16 pixels x all 64 out-channels: 4 accumulators -> 4 WMMAs per
// A-fragment (A loaded once from LDS, 4 B fragments streamed from L2-resident weights).
__global__ __launch_bounds__(128) void k_fus1(
    const _Float16* __restrict__ catp, const _Float16* __restrict__ wB,
    const float* __restrict__ bias, _Float16* __restrict__ hout) {
  extern __shared__ _Float16 smA[];           // [3][66][192]
  int g = blockIdx.x;
  int b = g >> 12;                            // 4096 row-groups per image
  int rem = g & 4095;
  int y = rem >> 3;
  int x0 = (rem & 7) << 6;

  // cooperative halo fill: 3*66*24 uint4 chunks (catp rows y..y+2, cols x0..x0+65, padded)
  for (int q = threadIdx.x; q < 3 * 66 * 24; q += 128) {
    int j = q % 24;
    int pc = q / 24;
    int cx = pc % 66;
    int ry = pc / 66;
    const uint4* src =
        (const uint4*)(catp + (((size_t)b * 514 + (y + ry)) * 514 + (x0 + cx)) * 192) + j;
    ((uint4*)smA)[q] = *src;
  }
  __syncthreads();

  int lane = threadIdx.x & 31;
  int wave = threadIdx.x >> 5;
  int mrow = lane & 15;
  int hi = lane >> 4;
  int kba = hi << 3, kbb = hi << 4;

  const _Float16* brow[4];
#pragma unroll
  for (int nt = 0; nt < 4; ++nt) brow[nt] = wB + (size_t)(nt * 16 + mrow) * cfg::KF1;

  v8f acc[4] = {};
  for (int ky = 0; ky < 3; ++ky) {
    for (int kx = 0; kx < 3; ++kx) {
      const _Float16* ap0 = smA + ((size_t)(ky * 66) + (wave * 16 + mrow + kx)) * 192;
      int tapoff = (ky * 3 + kx) * 192;
      for (int c0 = 0; c0 < 192; c0 += 32) {
        v16h af;
        const _Float16* ap = ap0 + c0;
#pragma unroll
        for (int j = 0; j < 16; ++j) af[j] = ap[kbb + j];
#pragma unroll
        for (int nt = 0; nt < 4; ++nt) {
          v16h bf;
          const _Float16* bp = brow[nt] + tapoff + c0 + kbb;
#pragma unroll
          for (int j = 0; j < 16; ++j) bf[j] = bp[j];
          acc[nt] = __builtin_amdgcn_wmma_f32_16x16x32_f16(false, af, false, bf,
                                                           (short)0, acc[nt], false, false);
        }
      }
    }
  }

#pragma unroll
  for (int nt = 0; nt < 4; ++nt) {
    int n = nt * 16 + mrow;
    float bv = bias[n];
#pragma unroll
    for (int r = 0; r < 8; ++r) {
      int xp = x0 + wave * 16 + (hi << 3) + r;
      hout[(((size_t)b * 512 + y) * 512 + xp) * 64 + n] = (_Float16)geluf(acc[nt][r] + bv);
    }
  }
}

// ---------------- 1x1 conv to K (3ch) + dehaze formula, NCHW out ----------------
__global__ void k_fus2_final(const _Float16* __restrict__ h, const float* __restrict__ w,
                             const float* __restrict__ bias, const float* __restrict__ hazy,
                             float* __restrict__ out) {
  size_t total = (size_t)cfg::NB * 3 * 512 * 512;
  size_t idx = (size_t)blockIdx.x * 256 + threadIdx.x;
  if (idx >= total) return;
  int x = idx % 512;
  int y = (idx / 512) % 512;
  int o = (idx / (512 * 512)) % 3;
  int b = idx / ((size_t)3 * 512 * 512);
  const _Float16* hp = h + ((size_t)(b * 512 + y) * 512 + x) * 64;
  float s = bias[o];
  for (int c = 0; c < 64; ++c) s += (float)hp[c] * w[o * 64 + c];
  float hz = hazy[idx];
  float v = s * hz - s + 1.f;
  out[idx] = fminf(fmaxf(v, 0.f), 1.f);
}

// =====================================================================
// Host orchestration
// =====================================================================
extern "C" void kernel_launch(void* const* d_in, const int* in_sizes, int n_in,
                              void* d_out, int out_size, void* d_ws, size_t ws_size,
                              hipStream_t stream) {
  (void)in_sizes; (void)n_in; (void)out_size; (void)ws_size;
  using namespace cfg;

  const float* hazy = (const float*)d_in[0];
  auto P = [&](int i) { return (const float*)d_in[i]; };

  // params pytree flattens with sorted dict keys:
  //   blocks[4] first, each block: bwd(9), fusion_b, fusion_w, fwd(9), ln_b, ln_g
  //   per-mamba sorted: A_log, D, conv_b, conv_w, dt_proj_b, dt_proj_w, in_proj_w, out_proj_w, x_proj_w
  auto blkBase = [](int blk) { return 1 + blk * 22; };
  const int I_ALOG = 0, I_D = 1, I_CB = 2, I_CW = 3, I_DTB = 4, I_DTW = 5,
            I_INW = 6, I_OUTW = 7, I_XPW = 8;
  const int I_FUSB = 9, I_FUSW = 10, I_FWD = 11, I_LNB = 20, I_LNG = 21;
  const int G = 1 + NLAY * 22;  // 89
  const int i_fus1_b = G + 0, i_fus1_w = G + 1, i_fus2_b = G + 2, i_fus2_w = G + 3,
            i_pe_b = G + 4, i_pe_w = G + 5, i_pos = G + 6, i_proj_b = G + 7,
            i_proj_w = G + 8, i_s16_b = G + 9, i_s16_w = G + 10, i_s1_b = G + 11,
            i_s1_w = G + 12, i_s2_b = G + 13, i_s2_w = G + 14, i_s4_b = G + 15,
            i_s4_w = G + 16, i_s8_b = G + 17, i_s8_w = G + 18;

  // ---- scratch layout ----
  uint8_t* ws = (uint8_t*)d_ws;
  size_t off = 0;
  auto alloc = [&](size_t bytes) -> void* {
    void* p = ws + off;
    off = (off + bytes + 255) & ~(size_t)255;
    return p;
  };
  _Float16* a_pe  = (_Float16*)alloc((size_t)NT * 192 * 2);
  _Float16* wslot = (_Float16*)alloc(262144);            // reused weight slot (max 64*1728 f16)
  float*    xA    = (float*)alloc((size_t)NT * 64 * 4);
  float*    xB    = (float*)alloc((size_t)NT * 64 * 4);
  _Float16* aw    = (_Float16*)alloc((size_t)NT * 64 * 2);
  _Float16* xzb[2] = { (_Float16*)alloc((size_t)NT * 256 * 2),
                       (_Float16*)alloc((size_t)NT * 256 * 2) };
  _Float16* ub[2]  = { (_Float16*)alloc((size_t)NT * 128 * 2),
                       (_Float16*)alloc((size_t)NT * 128 * 2) };
  float*    dblb[2] = { (float*)alloc((size_t)NT * 48 * 4),
                        (float*)alloc((size_t)NT * 48 * 4) };
  _Float16* yb[2]  = { (_Float16*)alloc((size_t)NT * 128 * 2),
                       (_Float16*)alloc((size_t)NT * 128 * 2) };
  _Float16* ycat  = (_Float16*)alloc((size_t)NT * 128 * 2);
  float*    yfus  = (float*)alloc((size_t)NT * 64 * 4);
  float*    xd    = (float*)alloc((size_t)NT * 64 * 4);
  float*    f16m  = (float*)alloc((size_t)NB * 32 * 32 * 64 * 4);
  float*    f8m   = (float*)alloc((size_t)NB * 64 * 64 * 64 * 4);
  float*    f4m   = (float*)alloc((size_t)NB * 128 * 128 * 32 * 4);
  float*    f2m   = (float*)alloc((size_t)NB * 256 * 256 * 16 * 4);
  float*    f1m   = (float*)alloc((size_t)NB * 512 * 512 * 16 * 4);
  size_t catpBytes = (size_t)NB * 514 * 514 * 192 * 2;
  _Float16* catp  = (_Float16*)alloc(catpBytes);
  _Float16* hbuf  = (_Float16*)alloc((size_t)NB * 512 * 512 * 64 * 2);

  auto egrid = [](size_t n) { return dim3((unsigned)((n + 255) / 256)); };
  auto cast2 = [&](const float* src, _Float16* dst, int sr, int sc, int dr, int dc) {
    k_cast_pad2<<<egrid((size_t)dr * dc), 256, 0, stream>>>(src, dst, sr, sc, dr, dc);
  };
  auto gemm = [&](const _Float16* A, int lda, const _Float16* W, int ldb,
                  const float* bias, float* Cf, _Float16* Ch, int ldc,
                  int M, int N, int K, int act) {
    dim3 g(N / 16, M / 64);
    k_gemm<<<g, dim3(128), 0, stream>>>(A, lda, W, ldb, bias, Cf, Ch, ldc, M, N, K, act);
  };

  // ---- patch embed + pos ----
  k_im2col<<<egrid((size_t)NT * 192), 256, 0, stream>>>(hazy, a_pe);
  cast2(P(i_pe_w), wslot, 64, 192, 64, 192);
  gemm(a_pe, 192, wslot, 192, P(i_pe_b), xA, nullptr, 64, NT, 64, 192, 0);
  k_add_pos<<<egrid((size_t)NT * 64), 256, 0, stream>>>(xA, P(i_pos));

  // ---- mamba blocks ----
  float* xcur = xA;
  float* xnext = xB;
  for (int blk = 0; blk < NLAY; ++blk) {
    int shift = (blk & 1) ? (WSZ / 2) : 0;
    int base = blkBase(blk);
    k_win_part<<<egrid((size_t)NT * 64), 256, 0, stream>>>(xcur, aw, shift);

    for (int d = 0; d < 2; ++d) {        // d=0 fwd, d=1 bwd
      int pb = base + (d == 0 ? I_FWD : 0);
      int dir = (d == 0) ? 1 : -1;
      cast2(P(pb + I_INW), wslot, 256, 64, 256, 64);
      gemm(aw, 64, wslot, 64, nullptr, nullptr, xzb[d], 256, NT, 256, 64, 0);
      k_dwconv<<<egrid((size_t)NWIN * DI), 256, 0, stream>>>(
          xzb[d], P(pb + I_CW), P(pb + I_CB), ub[d], dir);
      cast2(P(pb + I_XPW), wslot, 36, 128, 48, 128);
      gemm(ub[d], 128, wslot, 128, nullptr, dblb[d], nullptr, 48, NT, 48, 128, 0);
      k_scan<<<egrid((size_t)NWIN * DI), 256, 0, stream>>>(
          dblb[d], ub[d], xzb[d], P(pb + I_DTW), P(pb + I_DTB),
          P(pb + I_ALOG), P(pb + I_D), yb[d], dir);
      cast2(P(pb + I_OUTW), wslot, 64, 128, 64, 128);
      gemm(yb[d], 128, wslot, 128, nullptr, nullptr, ycat + d * 64, 128, NT, 64, 128, 0);
    }
    cast2(P(base + I_FUSW), wslot, 64, 128, 64, 128);
    gemm(ycat, 128, wslot, 128, P(base + I_FUSB), yfus, nullptr, 64, NT, 64, 128, 0);
    k_unwin_ln<<<egrid((size_t)NT), 256, 0, stream>>>(
        xcur, yfus, P(base + I_LNG), P(base + I_LNB), xnext, shift);
    float* tmp = xcur; xcur = xnext; xnext = tmp;
  }

  // ---- decoder ----
  cast2(xcur, aw, NT, 64, NT, 64);
  cast2(P(i_proj_w), wslot, 64, 64, 64, 64);
  gemm(aw, 64, wslot, 64, P(i_proj_b), xd, nullptr, 64, NT, 64, 64, 0);

  k_conv3x3<<<egrid((size_t)NB * 32 * 32 * 64), 256, 0, stream>>>(
      xd, P(i_s16_w), P(i_s16_b), f16m, NB, 64, 64, 64, 64, 2);
  k_conv3x3<<<egrid((size_t)NB * 64 * 64 * 64), 256, 0, stream>>>(
      xd, P(i_s8_w), P(i_s8_b), f8m, NB, 64, 64, 64, 64, 1);
  k_convT2<<<egrid((size_t)NB * 128 * 128 * 32), 256, 0, stream>>>(
      xd, P(i_s4_w), P(i_s4_b), f4m, NB, 64, 64, 64, 32);
  k_convT2<<<egrid((size_t)NB * 256 * 256 * 16), 256, 0, stream>>>(
      f4m, P(i_s2_w), P(i_s2_b), f2m, NB, 128, 128, 32, 16);
  k_convT2<<<egrid((size_t)NB * 512 * 512 * 16), 256, 0, stream>>>(
      f2m, P(i_s1_w), P(i_s1_b), f1m, NB, 256, 256, 16, 16);

  hipMemsetAsync(catp, 0, catpBytes, stream);
  k_bilinear<<<egrid((size_t)NB * 512 * 512 * 64), 256, 0, stream>>>(f16m, NB, 32, 32, 64, catp, 0);
  k_bilinear<<<egrid((size_t)NB * 512 * 512 * 64), 256, 0, stream>>>(f8m, NB, 64, 64, 64, catp, 64);
  k_bilinear<<<egrid((size_t)NB * 512 * 512 * 32), 256, 0, stream>>>(f4m, NB, 128, 128, 32, catp, 128);
  k_bilinear<<<egrid((size_t)NB * 512 * 512 * 16), 256, 0, stream>>>(f2m, NB, 256, 256, 16, catp, 160);
  k_bilinear<<<egrid((size_t)NB * 512 * 512 * 16), 256, 0, stream>>>(f1m, NB, 512, 512, 16, catp, 176);

  k_pack_fus1<<<egrid((size_t)64 * KF1), 256, 0, stream>>>(P(i_fus1_w), wslot);
  k_fus1<<<dim3((unsigned)((size_t)NB * 512 * 512 / 64)), dim3(128),
           3 * 66 * 192 * sizeof(_Float16), stream>>>(catp, wslot, P(i_fus1_b), hbuf);
  k_fus2_final<<<egrid((size_t)NB * 3 * 512 * 512), 256, 0, stream>>>(
      hbuf, P(i_fus2_w), P(i_fus2_b), hazy, (float*)d_out);
}
